// FlowBlock_60241211293819
// MI455X (gfx1250) — matrix operand order repaired and checked by reading, not compiled, not run on hardware
//
#include <hip/hip_runtime.h>
#include <hip/hip_bf16.h>
#include <math.h>

typedef __attribute__((ext_vector_type(16))) _Float16 v16h;
typedef __attribute__((ext_vector_type(8)))  _Float16 v8h;
typedef __attribute__((ext_vector_type(8)))  float    v8f;

#define PTS     8192          // B*N points
#define NB      4             // batches
#define NPB     2048          // points per batch
#define HD      512           // hidden dim
#define CD      64            // context dim
#define EPS_BN  1e-4f
#define PLANE   ((size_t)PTS * HD)   // one activation plane (primal or one tangent)
#define BS_LD   72            // LDS B-panel row stride in halves (144B: 16B aligned, low conflict)

// ---------------------------------------------------------------------------
// Packing / conversion kernels
// ---------------------------------------------------------------------------

// Wp[k*N + n] = (f16) W[n*K + k]   (transpose (out,in) weights into K-major for WMMA B)
__global__ void pack_w_kernel(const float* __restrict__ W, _Float16* __restrict__ Wp,
                              int N, int K) {
    int i = blockIdx.x * blockDim.x + threadIdx.x;
    if (i >= N * K) return;
    int k = i / N, n = i % N;
    Wp[(size_t)k * N + n] = (_Float16)W[(size_t)n * K + k];
}

// ctx weights (out x 65): drop t-column, pack Wp[k*N+n] = Wg[n*65 + 1 + k], k in [0,64)
__global__ void pack_ctx_kernel(const float* __restrict__ Wg, _Float16* __restrict__ Wp, int N) {
    int i = blockIdx.x * blockDim.x + threadIdx.x;
    if (i >= CD * N) return;
    int k = i / N, n = i % N;
    Wp[(size_t)k * N + n] = (_Float16)Wg[(size_t)n * 65 + 1 + k];
}

__global__ void conv_f16_kernel(const float* __restrict__ src, _Float16* __restrict__ dst, int n) {
    int i = blockIdx.x * blockDim.x + threadIdx.x;
    if (i < n) dst[i] = (_Float16)src[i];
}

// ---------------------------------------------------------------------------
// f16 WMMA GEMM:  C[M x N] = A[M x K] * Bp[K x N]   (f16 in, f16 out, f32 acc)
// Block: 256 thr = 8 waves. B panel (K x 64) staged once in LDS, shared by all
// waves. Each wave computes a 32-row x 64-col tile: 2 A fragments x 4 N tiles
// = 8 WMMAs per k-step.   grid = (M/256, N/64)
// ---------------------------------------------------------------------------
__global__ __launch_bounds__(256)
void gemm_f16_kernel(const _Float16* __restrict__ A, const _Float16* __restrict__ Bp,
                     _Float16* __restrict__ C, int M, int N, int K) {
    __shared__ __align__(16) _Float16 Bs[HD * BS_LD];   // max K=512: 72KB

    const int lane = threadIdx.x & 31;
    const int wave = threadIdx.x >> 5;
    const int mtile = blockIdx.x * 256 + wave * 32;
    const int ncol0 = blockIdx.y * 64;

    // ---- cooperative stage of B panel: Bs[k][j] = Bp[k][ncol0 + j], j in [0,64) ----
    for (int cidx = threadIdx.x; cidx < K * 8; cidx += 256) {
        int kk = cidx >> 3;
        int j0 = (cidx & 7) * 8;
        *(v8h*)&Bs[kk * BS_LD + j0] = *(const v8h*)(Bp + (size_t)kk * N + ncol0 + j0);
    }
    __syncthreads();

    const int g = lane >> 4;                       // lane group (K-half select for A)
    const _Float16* arow0 = A + (size_t)(mtile + (lane & 15)) * K + g * 8;
    const _Float16* arow1 = arow0 + (size_t)16 * K;

    v8f acc0[4] = {{}, {}, {}, {}};
    v8f acc1[4] = {{}, {}, {}, {}};

    for (int k = 0; k < K; k += 32) {
        // A fragments: 16x32 each; elems 0-7 -> K=k+8g+0..7, elems 8-15 -> K=k+16+8g+0..7
        v8h a00 = *(const v8h*)(arow0 + k);
        v8h a01 = *(const v8h*)(arow0 + k + 16);
        v8h a10 = *(const v8h*)(arow1 + k);
        v8h a11 = *(const v8h*)(arow1 + k + 16);
        v16h av0, av1;
#pragma unroll
        for (int e = 0; e < 8; ++e) {
            av0[e] = a00[e]; av0[e + 8] = a01[e];
            av1[e] = a10[e]; av1[e + 8] = a11[e];
        }

        // B fragments from LDS: lane holds N = lane&15; elems 0-7 -> K=k+8g+0..7, elems 8-15 -> K=k+16+8g+0..7
        const _Float16* bl = &Bs[(k + 8 * g) * BS_LD + (lane & 15)];
#pragma unroll
        for (int t = 0; t < 4; ++t) {
            v16h bv;
#pragma unroll
            for (int e = 0; e < 8; ++e) { bv[e] = bl[e * BS_LD + t * 16]; bv[e + 8] = bl[(e + 16) * BS_LD + t * 16]; }
            acc0[t] = __builtin_amdgcn_wmma_f32_16x16x32_f16(false, av0, false, bv, (short)0, acc0[t], false, false);
            acc1[t] = __builtin_amdgcn_wmma_f32_16x16x32_f16(false, av1, false, bv, (short)0, acc1[t], false, false);
        }
    }

    // C/D layout: lane -> n = base + (lane&15); VGPR r -> m = tile + r + 8*(lane>>4)
    const int nn = lane & 15;
#pragma unroll
    for (int r = 0; r < 8; ++r) {
        size_t row0 = (size_t)(mtile + r + 8 * g) * N + ncol0;
        size_t row1 = row0 + (size_t)16 * N;
#pragma unroll
        for (int t = 0; t < 4; ++t) {
            C[row0 + nn + t * 16] = (_Float16)acc0[t][r];
            C[row1 + nn + t * 16] = (_Float16)acc1[t][r];
        }
    }
}

// ---------------------------------------------------------------------------
// Layer-3 ctx precompute: g3pre[p][j] = sum_k c[p][k]*Wg3[j][1+k];  b3c likewise
// ---------------------------------------------------------------------------
__global__ void g3_kernel(const float* __restrict__ c, const float* __restrict__ Wg3,
                          const float* __restrict__ Wb3, float* __restrict__ g3pre,
                          float* __restrict__ b3c) {
    int i = blockIdx.x * blockDim.x + threadIdx.x;
    if (i >= PTS * 3) return;
    int p = i / 3, j = i % 3;
    float sg = 0.f, sb = 0.f;
    const float* cp = c + (size_t)p * CD;
#pragma unroll 8
    for (int k = 0; k < CD; ++k) {
        sg += cp[k] * Wg3[j * 65 + 1 + k];
        sb += cp[k] * Wb3[j * 65 + 1 + k];
    }
    g3pre[i] = sg; b3c[i] = sb;
}

// ---------------------------------------------------------------------------
// Init: x0 = xcur = BN1(x); Lint = 0
// ---------------------------------------------------------------------------
__global__ void init_kernel(const float* __restrict__ x, const float* __restrict__ m1,
                            const float* __restrict__ v1, const float* __restrict__ lg1,
                            const float* __restrict__ be1, float* __restrict__ x0,
                            float* __restrict__ xcur, float* __restrict__ Lint) {
    int p = blockIdx.x * blockDim.x + threadIdx.x;
    if (p >= PTS) return;
#pragma unroll
    for (int d = 0; d < 3; ++d) {
        float z = (x[p * 3 + d] - m1[d]) * expf(lg1[d]) * rsqrtf(v1[d] + EPS_BN) + be1[d];
        x0[p * 3 + d] = z; xcur[p * 3 + d] = z;
    }
    Lint[p] = 0.f;
}

__device__ __forceinline__ float stage_t(const float* sqrtT, int step, int stage, float* dt_out) {
    float T = sqrtT[0] * sqrtT[0];
    float dt = T / 3.0f;
    float fr = (stage == 0) ? 0.0f : ((stage == 3) ? 1.0f : 0.5f);
    *dt_out = dt;
    return ((float)step + fr) * dt;
}

// ---------------------------------------------------------------------------
// Layer 0 (K=3 linear, done elementwise) + gate/bias/tanh + tangent seed
// a planes: 0 = primal, 1..3 = tangents
// ---------------------------------------------------------------------------
__global__ __launch_bounds__(256)
void layer0_kernel(const float* __restrict__ xcur, const float* __restrict__ W0,
                   const float* __restrict__ b0, const _Float16* __restrict__ Gpre,
                   const _Float16* __restrict__ BiasC, const float* __restrict__ Wg,
                   const float* __restrict__ bg, const float* __restrict__ Wb,
                   const float* __restrict__ sqrtT, int step, int stage,
                   _Float16* __restrict__ a) {
    int idx = blockIdx.x * blockDim.x + threadIdx.x;       // P*HD threads
    int p = idx >> 9, n = idx & (HD - 1);
    float dt; float t = stage_t(sqrtT, step, stage, &dt);

    float w0 = W0[n * 3 + 0], w1 = W0[n * 3 + 1], w2 = W0[n * 3 + 2];
    float uP = xcur[p * 3 + 0] * w0 + xcur[p * 3 + 1] * w1 + xcur[p * 3 + 2] * w2 + b0[n];

    float gpre = (float)Gpre[idx] + t * Wg[n * 65] + bg[n];
    float gate = 1.0f / (1.0f + expf(-gpre));
    float bv   = (float)BiasC[idx] + t * Wb[n * 65];

    float aP = tanhf(uP * gate + bv);
    float d  = gate * (1.0f - aP * aP);

    a[idx]             = (_Float16)aP;
    a[PLANE     + idx] = (_Float16)(w0 * d);
    a[2 * PLANE + idx] = (_Float16)(w1 * d);
    a[3 * PLANE + idx] = (_Float16)(w2 * d);
}

// ---------------------------------------------------------------------------
// Epilogue for inner layers (1 and 2): primal h=(u+b)*gate+bias, a=tanh(h),
// d = gate*(1-a^2); tangents a_t = u_t * d. Reads u planes, rewrites a planes.
// ---------------------------------------------------------------------------
__global__ __launch_bounds__(256)
void epilogue_kernel(const _Float16* __restrict__ u, const float* __restrict__ b,
                     const _Float16* __restrict__ Gpre, const _Float16* __restrict__ BiasC,
                     const float* __restrict__ Wg, const float* __restrict__ bg,
                     const float* __restrict__ Wb, const float* __restrict__ sqrtT,
                     int step, int stage, _Float16* __restrict__ a) {
    int idx = blockIdx.x * blockDim.x + threadIdx.x;       // P*HD threads
    int n = idx & (HD - 1);
    float dt; float t = stage_t(sqrtT, step, stage, &dt);

    float gpre = (float)Gpre[idx] + t * Wg[n * 65] + bg[n];
    float gate = 1.0f / (1.0f + expf(-gpre));
    float bv   = (float)BiasC[idx] + t * Wb[n * 65];

    float aP = tanhf(((float)u[idx] + b[n]) * gate + bv);
    float d  = gate * (1.0f - aP * aP);

    a[idx]             = (_Float16)aP;
    a[PLANE     + idx] = (_Float16)((float)u[PLANE     + idx] * d);
    a[2 * PLANE + idx] = (_Float16)((float)u[2 * PLANE + idx] * d);
    a[3 * PLANE + idx] = (_Float16)((float)u[3 * PLANE + idx] * d);
}

// ---------------------------------------------------------------------------
// Layer 3 (N=3) + divergence + RK4 stage accumulate.  One wave per point.
// ---------------------------------------------------------------------------
__global__ __launch_bounds__(256)
void layer3_rk4_kernel(const _Float16* __restrict__ a, const float* __restrict__ W3,
                       const float* __restrict__ b3, const float* __restrict__ Wg3,
                       const float* __restrict__ bg3, const float* __restrict__ Wb3,
                       const float* __restrict__ g3pre, const float* __restrict__ b3c,
                       const float* __restrict__ sqrtT, int step, int stage,
                       float* __restrict__ x0, float* __restrict__ xcur,
                       float* __restrict__ xacc, float* __restrict__ Lint,
                       float* __restrict__ Lacc) {
    int gid  = blockIdx.x * blockDim.x + threadIdx.x;
    int p    = gid >> 5;
    int lane = gid & 31;
    float dt; float t = stage_t(sqrtT, step, stage, &dt);

    const _Float16* ap = a + (size_t)p * HD;
    float dx[3]; float div = 0.f;
#pragma unroll
    for (int j = 0; j < 3; ++j) {
        const float* wrow = W3 + j * HD;
        const _Float16* at = a + (size_t)(1 + j) * PLANE + (size_t)p * HD;
        float sP = 0.f, sT = 0.f;
        for (int k = lane; k < HD; k += 32) {
            float w = wrow[k];
            sP += (float)ap[k] * w;
            sT += (float)at[k] * w;
        }
#pragma unroll
        for (int off = 16; off > 0; off >>= 1) {
            sP += __shfl_xor(sP, off, 32);
            sT += __shfl_xor(sT, off, 32);
        }
        float gpre = g3pre[p * 3 + j] + t * Wg3[j * 65] + bg3[j];
        float gate = 1.0f / (1.0f + expf(-gpre));
        dx[j] = (sP + b3[j]) * gate + (b3c[p * 3 + j] + t * Wb3[j * 65]);
        div  += sT * gate;
    }

    if (lane == 0) {
        float kl = -div;
        if (stage == 0) {
#pragma unroll
            for (int j = 0; j < 3; ++j) {
                xacc[p * 3 + j] = dx[j];
                xcur[p * 3 + j] = x0[p * 3 + j] + 0.5f * dt * dx[j];
            }
            Lacc[p] = kl;
        } else if (stage == 1 || stage == 2) {
            float cs = (stage == 1) ? 0.5f * dt : dt;
#pragma unroll
            for (int j = 0; j < 3; ++j) {
                xacc[p * 3 + j] += 2.0f * dx[j];
                xcur[p * 3 + j] = x0[p * 3 + j] + cs * dx[j];
            }
            Lacc[p] += 2.0f * kl;
        } else {
            float dt6 = dt / 6.0f;
#pragma unroll
            for (int j = 0; j < 3; ++j) {
                float xn = x0[p * 3 + j] + dt6 * (xacc[p * 3 + j] + dx[j]);
                x0[p * 3 + j] = xn; xcur[p * 3 + j] = xn;
            }
            Lint[p] += dt6 * (Lacc[p] + kl);
        }
    }
}

// ---------------------------------------------------------------------------
// BN2 + x output
// ---------------------------------------------------------------------------
__global__ void bn2_out_kernel(const float* __restrict__ x0, const float* __restrict__ m2,
                               const float* __restrict__ v2, const float* __restrict__ lg2,
                               const float* __restrict__ be2, float* __restrict__ out) {
    int i = blockIdx.x * blockDim.x + threadIdx.x;
    if (i >= PTS * 3) return;
    int d = i % 3;
    out[i] = (x0[i] - m2[d]) * expf(lg2[d]) * rsqrtf(v2[d] + EPS_BN) + be2[d];
}

// log-prob reduction per batch:  lp[b] = sum_n Lint - Npb*(ld1+ld2)
__global__ void lp_reduce_kernel(const float* __restrict__ Lint, const float* __restrict__ v1,
                                 const float* __restrict__ lg1, const float* __restrict__ v2,
                                 const float* __restrict__ lg2, float* __restrict__ out) {
    __shared__ float sm[256];
    int b = blockIdx.x;
    float s = 0.f;
    for (int n = threadIdx.x; n < NPB; n += 256) s += Lint[b * NPB + n];
    sm[threadIdx.x] = s;
    __syncthreads();
    for (int o = 128; o > 0; o >>= 1) {
        if (threadIdx.x < o) sm[threadIdx.x] += sm[threadIdx.x + o];
        __syncthreads();
    }
    if (threadIdx.x == 0) {
        float ld = 0.f;
        for (int d = 0; d < 3; ++d)
            ld += (lg1[d] - 0.5f * logf(v1[d] + EPS_BN)) + (lg2[d] - 0.5f * logf(v2[d] + EPS_BN));
        out[b] = sm[0] - (float)NPB * ld;
    }
}

// ---------------------------------------------------------------------------
// Launcher
// ---------------------------------------------------------------------------
extern "C" void kernel_launch(void* const* d_in, const int* in_sizes, int n_in,
                              void* d_out, int out_size, void* d_ws, size_t ws_size,
                              hipStream_t stream) {
    const float* x     = (const float*)d_in[0];
    const float* c     = (const float*)d_in[1];
    const float* m1    = (const float*)d_in[2];
    const float* v1    = (const float*)d_in[3];
    const float* lg1   = (const float*)d_in[4];
    const float* be1   = (const float*)d_in[5];
    const float* m2    = (const float*)d_in[6];
    const float* v2    = (const float*)d_in[7];
    const float* lg2   = (const float*)d_in[8];
    const float* be2   = (const float*)d_in[9];
    const float* sqrtT = (const float*)d_in[10];
    const float* W[4]  = {(const float*)d_in[11], (const float*)d_in[16], (const float*)d_in[21], (const float*)d_in[26]};
    const float* bL[4] = {(const float*)d_in[12], (const float*)d_in[17], (const float*)d_in[22], (const float*)d_in[27]};
    const float* Wg[4] = {(const float*)d_in[13], (const float*)d_in[18], (const float*)d_in[23], (const float*)d_in[28]};
    const float* bg[4] = {(const float*)d_in[14], (const float*)d_in[19], (const float*)d_in[24], (const float*)d_in[29]};
    const float* Wb[4] = {(const float*)d_in[15], (const float*)d_in[20], (const float*)d_in[25], (const float*)d_in[30]};
    float* out = (float*)d_out;

    // ---- workspace carve ----
    char* w = (char*)d_ws; size_t off = 0;
    auto alloc = [&](size_t bytes) -> void* {
        void* p = w + off; off = (off + bytes + 255) & ~(size_t)255; return p;
    };
    _Float16* W1p   = (_Float16*)alloc((size_t)HD * HD * 2);
    _Float16* W2p   = (_Float16*)alloc((size_t)HD * HD * 2);
    _Float16* Wgc[3], *Wbc[3], *Gpre[3], *BiasC[3];
    for (int i = 0; i < 3; ++i) Wgc[i] = (_Float16*)alloc((size_t)CD * HD * 2);
    for (int i = 0; i < 3; ++i) Wbc[i] = (_Float16*)alloc((size_t)CD * HD * 2);
    _Float16* cH    = (_Float16*)alloc((size_t)PTS * CD * 2);
    for (int i = 0; i < 3; ++i) Gpre[i]  = (_Float16*)alloc(PLANE * 2);
    for (int i = 0; i < 3; ++i) BiasC[i] = (_Float16*)alloc(PLANE * 2);
    _Float16* aBuf  = (_Float16*)alloc(4 * PLANE * 2);
    _Float16* uBuf  = (_Float16*)alloc(4 * PLANE * 2);
    float* x0    = (float*)alloc((size_t)PTS * 3 * 4);
    float* xcur  = (float*)alloc((size_t)PTS * 3 * 4);
    float* xacc  = (float*)alloc((size_t)PTS * 3 * 4);
    float* Lint  = (float*)alloc((size_t)PTS * 4);
    float* Lacc  = (float*)alloc((size_t)PTS * 4);
    float* g3pre = (float*)alloc((size_t)PTS * 3 * 4);
    float* b3c   = (float*)alloc((size_t)PTS * 3 * 4);
    (void)ws_size; (void)in_sizes; (void)n_in; (void)out_size;

    // ---- weight packing / ctx precompute ----
    pack_w_kernel<<<(HD * HD + 255) / 256, 256, 0, stream>>>(W[1], W1p, HD, HD);
    pack_w_kernel<<<(HD * HD + 255) / 256, 256, 0, stream>>>(W[2], W2p, HD, HD);
    for (int i = 0; i < 3; ++i) {
        pack_ctx_kernel<<<(CD * HD + 255) / 256, 256, 0, stream>>>(Wg[i], Wgc[i], HD);
        pack_ctx_kernel<<<(CD * HD + 255) / 256, 256, 0, stream>>>(Wb[i], Wbc[i], HD);
    }
    conv_f16_kernel<<<(PTS * CD + 255) / 256, 256, 0, stream>>>(c, cH, PTS * CD);

    // ctx GEMMs (t-independent parts of gates/biases): M=PTS, K=CD, N=HD
    dim3 gctx(PTS / 256, HD / 64);
    for (int i = 0; i < 3; ++i) {
        gemm_f16_kernel<<<gctx, 256, 0, stream>>>(cH, Wgc[i], Gpre[i], PTS, HD, CD);
        gemm_f16_kernel<<<gctx, 256, 0, stream>>>(cH, Wbc[i], BiasC[i], PTS, HD, CD);
    }
    g3_kernel<<<(PTS * 3 + 255) / 256, 256, 0, stream>>>(c, Wg[3], Wb[3], g3pre, b3c);
    init_kernel<<<(PTS + 255) / 256, 256, 0, stream>>>(x, m1, v1, lg1, be1, x0, xcur, Lint);

    // ---- RK4 integration: 3 steps x 4 stages ----
    dim3 ggemm(4 * PTS / 256, HD / 64);   // M = 4*PTS (primal + 3 tangent planes)
    const int ELT = PTS * HD / 256;
    for (int step = 0; step < 3; ++step) {
        for (int stage = 0; stage < 4; ++stage) {
            layer0_kernel<<<ELT, 256, 0, stream>>>(xcur, W[0], bL[0], Gpre[0], BiasC[0],
                                                   Wg[0], bg[0], Wb[0], sqrtT, step, stage, aBuf);
            gemm_f16_kernel<<<ggemm, 256, 0, stream>>>(aBuf, W1p, uBuf, 4 * PTS, HD, HD);
            epilogue_kernel<<<ELT, 256, 0, stream>>>(uBuf, bL[1], Gpre[1], BiasC[1],
                                                     Wg[1], bg[1], Wb[1], sqrtT, step, stage, aBuf);
            gemm_f16_kernel<<<ggemm, 256, 0, stream>>>(aBuf, W2p, uBuf, 4 * PTS, HD, HD);
            epilogue_kernel<<<ELT, 256, 0, stream>>>(uBuf, bL[2], Gpre[2], BiasC[2],
                                                     Wg[2], bg[2], Wb[2], sqrtT, step, stage, aBuf);
            layer3_rk4_kernel<<<PTS * 32 / 256, 256, 0, stream>>>(aBuf, W[3], bL[3], Wg[3], bg[3], Wb[3],
                                                                  g3pre, b3c, sqrtT, step, stage,
                                                                  x0, xcur, xacc, Lint, Lacc);
        }
    }

    // ---- outputs ----
    bn2_out_kernel<<<(PTS * 3 + 255) / 256, 256, 0, stream>>>(x0, m2, v2, lg2, be2, out);
    lp_reduce_kernel<<<NB, 256, 0, stream>>>(Lint, v1, lg1, v2, lg2, out + PTS * 3);
}
